// GAT_83459804495959
// MI455X (gfx1250) — matrix operand order repaired and checked by reading, not compiled, not run on hardware
//
#include <hip/hip_runtime.h>
#include <hip/hip_bf16.h>

// ---------------------------------------------------------------------------
// GAT (2-layer, PyG-style) for MI455X / gfx1250.
//   Layer1: h1 = x@W1 (WMMA f16->f32), per-head attention softmax over edges
//           (atomic max/exp-sum/scatter), bias + ELU.
//   Layer2: tiny [64->2] linear (VALU), 1-head attention, log_softmax.
// ---------------------------------------------------------------------------

typedef _Float16 v8h  __attribute__((ext_vector_type(8)));
typedef _Float16 v16h __attribute__((ext_vector_type(16)));
typedef float    v8f  __attribute__((ext_vector_type(8)));

#define F_IN   128
#define HID1   64     // HEADS*HIDDEN
#define HEADS1 8
#define CH1    8
#define OUT2   2
#define NEG_SLOPE 0.2f

// ------------------------- utility fill ------------------------------------
__global__ void fill_f32(float* __restrict__ p, float v, int n) {
    int i = blockIdx.x * blockDim.x + threadIdx.x;
    if (i < n) p[i] = v;
}

// ------------------- Layer-1 GEMM: h1 = x @ W1 via WMMA --------------------
// Block: 256 threads = 8 waves. Each block does 128 rows x 64 cols, K=128.
// Wave w computes rows [w*16, w*16+16) x all 64 cols (4 col tiles), 4 k-steps.
#define XLDS_STRIDE 136   // 128 + 8 halves: break 256B bank-conflict stride
__global__ __launch_bounds__(256) void gemm1_wmma(
    const float* __restrict__ x, const float* __restrict__ W,
    float* __restrict__ h1, int nnodes)
{
    __shared__ __align__(16) _Float16 sX[128 * XLDS_STRIDE]; // [row][k]
    __shared__ __align__(16) _Float16 sW[HID1 * XLDS_STRIDE]; // [n][k] (W^T)
    const int t = threadIdx.x;
    const int row0 = blockIdx.x * 128;

    // Stage W1 transposed (f32 -> f16): W[k*64+n] -> sW[n][k]
    for (int i = t; i < F_IN * HID1; i += 256) {
        int k = i >> 6, n = i & 63;
        sW[n * XLDS_STRIDE + k] = (_Float16)W[i];
    }
    // Stage x block (f32 -> f16), zero-pad past end of node array
    for (int i = t; i < 128 * F_IN; i += 256) {
        int r = i >> 7, k = i & 127;
        int gr = row0 + r;
        sX[r * XLDS_STRIDE + k] =
            (_Float16)(gr < nnodes ? x[(size_t)gr * F_IN + k] : 0.0f);
    }
    __syncthreads();

    const int wave = t >> 5;
    const int lane = t & 31;
    const int lrow = lane & 15;
    const int hi   = (lane & 16) ? 1 : 0;
    const int rowbase = wave * 16;

    v8f acc[4] = {};
    #pragma unroll
    for (int kt = 0; kt < 4; ++kt) {
        // A fragment 16x32 f16: lanes 0-15 hold K {0..7,16..23}, lanes 16-31 {8..15,24..31}
        union { v16h v; v8h h[2]; } a;
        const int arow = rowbase + lrow;
        const int kba = kt * 32 + (hi ? 8 : 0);
        a.h[0] = *(const v8h*)&sX[arow * XLDS_STRIDE + kba];
        a.h[1] = *(const v8h*)&sX[arow * XLDS_STRIDE + kba + 16];
        #pragma unroll
        for (int nt = 0; nt < 4; ++nt) {
            // B fragment 32x16 f16: lanes 0-15 col=n, K {0..7,16..23}; lanes 16-31 K {8..15,24..31}
            union { v16h v; v8h h[2]; } b;
            const int col = nt * 16 + lrow;
            const int kbb = kt * 32 + (hi ? 8 : 0);
            b.h[0] = *(const v8h*)&sW[col * XLDS_STRIDE + kbb];
            b.h[1] = *(const v8h*)&sW[col * XLDS_STRIDE + kbb + 16];
            acc[nt] = __builtin_amdgcn_wmma_f32_16x16x32_f16(
                false, a.v, false, b.v, (short)0, acc[nt], false, false);
        }
    }
    // C/D layout: VGPR r -> (M = r + 8*hi, N = lane&15)
    const int mofs = hi ? 8 : 0;
    #pragma unroll
    for (int nt = 0; nt < 4; ++nt) {
        const int col = nt * 16 + lrow;
        #pragma unroll
        for (int r = 0; r < 8; ++r) {
            int grow = row0 + rowbase + mofs + r;
            if (grow < nnodes) h1[(size_t)grow * HID1 + col] = acc[nt][r];
        }
    }
}

// ---------------- per-(node,head) attention scalars (layer 1) --------------
__global__ void alphas1(const float* __restrict__ h1,
                        const float* __restrict__ att_src,
                        const float* __restrict__ att_dst,
                        float* __restrict__ as, float* __restrict__ ad,
                        int nnodes)
{
    int i = blockIdx.x * blockDim.x + threadIdx.x;      // i = n*HEADS + h
    if (i >= nnodes * HEADS1) return;
    int n = i >> 3, h = i & 7;
    const float* row = h1 + (size_t)n * HID1 + h * CH1;
    float s = 0.f, d = 0.f;
    #pragma unroll
    for (int c = 0; c < CH1; ++c) {
        float v = row[c];
        s += v * att_src[h * CH1 + c];
        d += v * att_dst[h * CH1 + c];
    }
    as[i] = s;
    ad[i] = d;
}

// --------------------------- edge helpers ----------------------------------
__device__ __forceinline__ void edge_decode(int e, const int* __restrict__ ei,
                                            int nedges, int& src, int& dst) {
    if (e < nedges) { src = ei[e]; dst = ei[nedges + e]; }
    else            { src = dst = e - nedges; }           // self loop
}

__device__ __forceinline__ float leaky(float v) {
    return v > 0.f ? v : NEG_SLOPE * v;
}

// native CDNA5 float atomic max (no return -> STOREcnt; S_ENDPGM waits idle)
__device__ __forceinline__ void atomic_max_f32(float* p, float v) {
    unsigned long long addr = (unsigned long long)p;
    asm volatile("global_atomic_max_num_f32 %0, %1, off"
                 :: "v"(addr), "v"(v) : "memory");
}

// pass 1: m[dst,h] = segment-max of leaky(as[src,h] + ad[dst,h])
template <int H>
__global__ void edge_max(const int* __restrict__ ei, int nedges, int etot,
                         const float* __restrict__ as, const float* __restrict__ ad,
                         float* __restrict__ m)
{
    int gid = blockIdx.x * blockDim.x + threadIdx.x;
    if (gid >= etot * H) return;
    int e = gid / H, h = gid % H;
    int src, dst; edge_decode(e, ei, nedges, src, dst);
    float v = leaky(as[src * H + h] + ad[dst * H + h]);
    atomic_max_f32(&m[dst * H + h], v);
}

// pass 2: denom[dst,h] += exp(e - m[dst,h])
template <int H>
__global__ void edge_expsum(const int* __restrict__ ei, int nedges, int etot,
                            const float* __restrict__ as, const float* __restrict__ ad,
                            const float* __restrict__ m, float* __restrict__ denom)
{
    int gid = blockIdx.x * blockDim.x + threadIdx.x;
    if (gid >= etot * H) return;
    int e = gid / H, h = gid % H;
    int src, dst; edge_decode(e, ei, nedges, src, dst);
    float v = leaky(as[src * H + h] + ad[dst * H + h]);
    float w = __expf(v - m[dst * H + h]);
    atomicAdd(&denom[dst * H + h], w);
}

// pass 3: agg[dst, h, :] += feat[src, h, :] * (exp(e-m)/denom)
template <int H, int C>
__global__ void edge_aggregate(const int* __restrict__ ei, int nedges, int etot,
                               const float* __restrict__ as, const float* __restrict__ ad,
                               const float* __restrict__ m, const float* __restrict__ denom,
                               const float* __restrict__ feat, float* __restrict__ agg)
{
    int gid = blockIdx.x * blockDim.x + threadIdx.x;
    if (gid >= etot * H) return;
    int e = gid / H, h = gid % H;
    int src, dst; edge_decode(e, ei, nedges, src, dst);
    float v = leaky(as[src * H + h] + ad[dst * H + h]);
    float w = __expf(v - m[dst * H + h]);
    float alpha = w / fmaxf(denom[dst * H + h], 1e-16f);
    const float* frow = feat + (size_t)src * (H * C) + h * C;
    float* arow = agg + (size_t)dst * (H * C) + h * C;
    #pragma unroll
    for (int c = 0; c < C; ++c)
        atomicAdd(&arow[c], frow[c] * alpha);
}

// ---------------- bias + ELU (layer-1 output, in place) --------------------
__global__ void bias_elu(float* __restrict__ f, const float* __restrict__ bias,
                         int total, int width)
{
    int i = blockIdx.x * blockDim.x + threadIdx.x;
    if (i >= total) return;
    float v = f[i] + bias[i % width];
    f[i] = v > 0.f ? v : (__expf(v) - 1.0f);
}

// ---------------- layer-2 linear [64->2] + attention scalars ---------------
__global__ void linear2(const float* __restrict__ f2, const float* __restrict__ W2,
                        const float* __restrict__ att_src, const float* __restrict__ att_dst,
                        float* __restrict__ t2, float* __restrict__ as2,
                        float* __restrict__ ad2, int nnodes)
{
    int n = blockIdx.x * blockDim.x + threadIdx.x;
    if (n >= nnodes) return;
    const float* row = f2 + (size_t)n * HID1;
    float a0 = 0.f, a1 = 0.f;
    #pragma unroll 8
    for (int j = 0; j < HID1; ++j) {
        float v = row[j];
        a0 += v * W2[j * OUT2 + 0];
        a1 += v * W2[j * OUT2 + 1];
    }
    t2[n * OUT2 + 0] = a0;
    t2[n * OUT2 + 1] = a1;
    as2[n] = a0 * att_src[0] + a1 * att_src[1];
    ad2[n] = a0 * att_dst[0] + a1 * att_dst[1];
}

// ---------------- final: bias + log_softmax over 2 classes -----------------
__global__ void final_lsm(const float* __restrict__ agg2, const float* __restrict__ bias,
                          float* __restrict__ out, int nnodes)
{
    int n = blockIdx.x * blockDim.x + threadIdx.x;
    if (n >= nnodes) return;
    float v0 = agg2[n * 2 + 0] + bias[0];
    float v1 = agg2[n * 2 + 1] + bias[1];
    float mx = fmaxf(v0, v1);
    float lse = mx + __logf(__expf(v0 - mx) + __expf(v1 - mx));
    out[n * 2 + 0] = v0 - lse;
    out[n * 2 + 1] = v1 - lse;
}

// ---------------------------------------------------------------------------
extern "C" void kernel_launch(void* const* d_in, const int* in_sizes, int n_in,
                              void* d_out, int out_size, void* d_ws, size_t ws_size,
                              hipStream_t stream)
{
    const float* x    = (const float*)d_in[0];
    const int*   ei   = (const int*)  d_in[1];
    const float* W1   = (const float*)d_in[2];
    const float* aS1  = (const float*)d_in[3];
    const float* aD1  = (const float*)d_in[4];
    const float* b1   = (const float*)d_in[5];
    const float* W2   = (const float*)d_in[6];
    const float* aS2  = (const float*)d_in[7];
    const float* aD2  = (const float*)d_in[8];
    const float* b2   = (const float*)d_in[9];

    const int nnodes = in_sizes[0] / F_IN;
    const int nedges = in_sizes[1] / 2;
    const int etot   = nedges + nnodes;          // with self loops
    (void)n_in; (void)out_size; (void)ws_size;

    // workspace layout (floats)
    float* ws   = (float*)d_ws;
    size_t off  = 0;
    float* h1   = ws + off; off += (size_t)nnodes * HID1;    // pre-agg features
    float* f2   = ws + off; off += (size_t)nnodes * HID1;    // agg1 -> elu output
    float* as1  = ws + off; off += (size_t)nnodes * HEADS1;
    float* ad1  = ws + off; off += (size_t)nnodes * HEADS1;
    float* m1   = ws + off; off += (size_t)nnodes * HEADS1;
    float* d1   = ws + off; off += (size_t)nnodes * HEADS1;
    float* t2   = ws + off; off += (size_t)nnodes * OUT2;
    float* as2  = ws + off; off += (size_t)nnodes;
    float* ad2  = ws + off; off += (size_t)nnodes;
    float* m2   = ws + off; off += (size_t)nnodes;
    float* d2   = ws + off; off += (size_t)nnodes;
    float* agg2 = ws + off; off += (size_t)nnodes * OUT2;
    float* out  = (float*)d_out;

    const int T = 256;
    #define GRID(n) dim3(((n) + T - 1) / T)
    const float NEG_INF = -3.0e38f;

    // deterministic re-init every call (graph replay safe)
    fill_f32<<<GRID(nnodes * HID1),   T, 0, stream>>>(f2,   0.f,     nnodes * HID1);
    fill_f32<<<GRID(nnodes * HEADS1), T, 0, stream>>>(d1,   0.f,     nnodes * HEADS1);
    fill_f32<<<GRID(nnodes * HEADS1), T, 0, stream>>>(m1,   NEG_INF, nnodes * HEADS1);
    fill_f32<<<GRID(nnodes * OUT2),   T, 0, stream>>>(agg2, 0.f,     nnodes * OUT2);
    fill_f32<<<GRID(nnodes),          T, 0, stream>>>(d2,   0.f,     nnodes);
    fill_f32<<<GRID(nnodes),          T, 0, stream>>>(m2,   NEG_INF, nnodes);

    // layer 1
    gemm1_wmma<<<dim3((nnodes + 127) / 128), T, 0, stream>>>(x, W1, h1, nnodes);
    alphas1<<<GRID(nnodes * HEADS1), T, 0, stream>>>(h1, aS1, aD1, as1, ad1, nnodes);
    edge_max<HEADS1>      <<<GRID(etot * HEADS1), T, 0, stream>>>(ei, nedges, etot, as1, ad1, m1);
    edge_expsum<HEADS1>   <<<GRID(etot * HEADS1), T, 0, stream>>>(ei, nedges, etot, as1, ad1, m1, d1);
    edge_aggregate<HEADS1, CH1><<<GRID(etot * HEADS1), T, 0, stream>>>(ei, nedges, etot, as1, ad1, m1, d1, h1, f2);
    bias_elu<<<GRID(nnodes * HID1), T, 0, stream>>>(f2, b1, nnodes * HID1, HID1);

    // layer 2
    linear2<<<GRID(nnodes), T, 0, stream>>>(f2, W2, aS2, aD2, t2, as2, ad2, nnodes);
    edge_max<1>      <<<GRID(etot), T, 0, stream>>>(ei, nedges, etot, as2, ad2, m2);
    edge_expsum<1>   <<<GRID(etot), T, 0, stream>>>(ei, nedges, etot, as2, ad2, m2, d2);
    edge_aggregate<1, OUT2><<<GRID(etot), T, 0, stream>>>(ei, nedges, etot, as2, ad2, m2, d2, t2, agg2);
    final_lsm<<<GRID(nnodes), T, 0, stream>>>(agg2, b2, out, nnodes);
    #undef GRID
}
